// DeltaNet_6614249636545
// MI455X (gfx1250) — matrix operand order repaired and checked by reading, not compiled, not run on hardware
//
#include <hip/hip_runtime.h>
#include <hip/hip_bf16.h>

// ---------------------------------------------------------------------------
// DeltaNet chunked forward for MI455X (gfx1250, wave32, WMMA 16x16x32 f16).
//  kernel 1: beta/g gate projections [8192x2048]@[2048x16]; weights staged
//            transposed as f16 in LDS, inner loop = ds reads + v_wmma.
//  kernel 2: one persistent workgroup per (b,h). f32 state master in VGPRs
//            (8 tiles/wave), f16 transposed mirror in LDS feeds WMMA.
//            o_proj weights DMA'd once via TDM (tensor_load_to_lds) and kept
//            as f16 in LDS. All chunk matmuls are v_wmma_f32_16x16x32_f16.
// ---------------------------------------------------------------------------

typedef __attribute__((ext_vector_type(16))) _Float16 v16h;
typedef __attribute__((ext_vector_type(8)))  float    v8f;
typedef unsigned int u32x4 __attribute__((ext_vector_type(4)));
typedef int          i32x4 __attribute__((ext_vector_type(4)));
typedef int          i32x8 __attribute__((ext_vector_type(8)));

#define WMMA_F16(a, b, c) \
  __builtin_amdgcn_wmma_f32_16x16x32_f16(false, (a), false, (b), (short)0, (c), false, false)

#if defined(__has_builtin)
#if __has_builtin(__builtin_amdgcn_tensor_load_to_lds)
#define DN_HAS_TDM 1
#endif
#endif

namespace dn {
constexpr int kB = 2, kT = 4096, kH = 16, kDK = 128, kDV = 128, kHID = 2048, kC = 64;
constexpr int kN = kT / kC;                     // 64 chunks
constexpr float kScale = 0.08838834764831845f;  // DK^-0.5
}

// ---- fragment helpers (CDNA5 wave32 WMMA layouts) -------------------------
__device__ __forceinline__ int amap_k(int i, int lane) {
  int base = (lane & 16) ? 16 : 0;
  return base + 2 * i;
}

__device__ __forceinline__ v16h fragA_lds(const _Float16* p, int ld, int m0, int k0, int lane) {
  v16h a; const int m = m0 + (lane & 15);
#pragma unroll
  for (int i = 0; i < 8; ++i) {
    const int k = k0 + amap_k(i, lane);
    a[2 * i]     = p[m * ld + k];
    a[2 * i + 1] = p[m * ld + k + 1];
  }
  return a;
}

// A = X^T where X is row-major f16 LDS:  A(m,k) = X[k][m]
__device__ __forceinline__ v16h fragAT_lds(const _Float16* p, int ld, int m0, int k0, int lane) {
  v16h a; const int m = m0 + (lane & 15);
#pragma unroll
  for (int i = 0; i < 8; ++i) {
    const int k = k0 + amap_k(i, lane);
    a[2 * i]     = p[k * ld + m];
    a[2 * i + 1] = p[(k + 1) * ld + m];
  }
  return a;
}

// B 32x16 f16, row-major source: B(k,n) = p[(k0+k)*ld + n0+n]
__device__ __forceinline__ v16h fragB_lds(const _Float16* p, int ld, int k0, int n0, int lane) {
  v16h b; const int n = n0 + (lane & 15);
  const int kk = k0 + ((lane & 16) ? 16 : 0);
#pragma unroll
  for (int e = 0; e < 16; ++e) b[e] = p[(kk + e) * ld + n];
  return b;
}

// B = X^T where X is row-major f16 LDS: B(k,n) = X[n][k]  (contiguous per lane)
__device__ __forceinline__ v16h fragBT_lds(const _Float16* p, int ld, int k0, int n0, int lane) {
  v16h b; const int n = n0 + (lane & 15);
  const _Float16* r = p + n * ld + k0 + ((lane & 16) ? 16 : 0);
#pragma unroll
  for (int e = 0; e < 16; ++e) b[e] = r[e];
  return b;
}

// A from row-major f32 global (gate-proj hidden states)
__device__ __forceinline__ v16h fragA_f32g(const float* p, int ld, int m0, int k0, int lane) {
  v16h a; const size_t m = (size_t)(m0 + (lane & 15));
#pragma unroll
  for (int i = 0; i < 8; ++i) {
    const int k = k0 + amap_k(i, lane);
    const float2 f = *reinterpret_cast<const float2*>(p + m * ld + k);
    a[2 * i]     = (_Float16)f.x;
    a[2 * i + 1] = (_Float16)f.y;
  }
  return a;
}

// C/D 16x16 f32: VGPR r -> M = r + (lane<16?0:8), N = lane&15
__device__ __forceinline__ void storeD_f16(_Float16* p, int ld, int m0, int n0, int lane, v8f d) {
  const int n = n0 + (lane & 15), mo = m0 + ((lane & 16) ? 8 : 0);
#pragma unroll
  for (int r = 0; r < 8; ++r) p[(mo + r) * ld + n] = (_Float16)d[r];
}
// transposed store: p[n][m] (used for the f16 state mirror ShT)
__device__ __forceinline__ void storeD_f16T(_Float16* p, int ld, int m0, int n0, int lane, v8f d) {
  const int n = n0 + (lane & 15), mo = m0 + ((lane & 16) ? 8 : 0);
#pragma unroll
  for (int r = 0; r < 8; ++r) p[n * ld + (mo + r)] = (_Float16)d[r];
}

// fast sigmoid: v_exp + v_rcp (no IEEE divide chains)
__device__ __forceinline__ float sigmoidf_(float x) {
  return __builtin_amdgcn_rcpf(1.0f + __expf(-x));
}

// ---- TDM: DMA a 2D f32 tile (rows x 128, row stride in elems) to LDS ------
__device__ __forceinline__ void tdm_load_2d_f32(unsigned lds_addr, const void* gaddr,
                                                int width, int rows, int stride) {
#if defined(DN_HAS_TDM) && defined(__HIP_DEVICE_COMPILE__)
  const unsigned long long ga = (unsigned long long)(uintptr_t)gaddr;
  u32x4 g0;
  g0[0] = 1u;                                                  // count=1, user D#
  g0[1] = lds_addr;                                            // LDS byte address
  g0[2] = (unsigned)(ga & 0xFFFFFFFFu);                        // global_addr[31:0]
  g0[3] = (unsigned)((ga >> 32) & 0x01FFFFFFu) | (2u << 30);   // addr hi | type=2
  i32x8 g1;
  g1[0] = 2 << 16;                                             // data_size = 4B
  g1[1] = (width & 0xFFFF) << 16;                              // tensor_dim0 lo16
  g1[2] = ((unsigned)width >> 16) | ((rows & 0xFFFF) << 16);   // dim0 hi | dim1 lo
  g1[3] = ((unsigned)rows >> 16) | ((width & 0xFFFF) << 16);   // dim1 hi | tile_dim0
  g1[4] = rows & 0xFFFF;                                       // tile_dim1, tile_dim2=0
  g1[5] = stride;                                              // tensor_dim0_stride lo32
  g1[6] = 0;
  g1[7] = 0;
  const i32x4 gz = {0, 0, 0, 0};
#if __clang_major__ >= 23
  const i32x8 gz8 = {0, 0, 0, 0, 0, 0, 0, 0};
  __builtin_amdgcn_tensor_load_to_lds(g0, g1, gz, gz, gz8, 0);
#else
  __builtin_amdgcn_tensor_load_to_lds(g0, g1, gz, gz, 0);
#endif
  __builtin_amdgcn_s_wait_tensorcnt((short)0);
#else
  (void)lds_addr; (void)gaddr; (void)width; (void)rows; (void)stride;
#endif
}

// ---------------------------------------------------------------------------
// Kernel 1: beta = sigmoid(hidden_ab @ Wb), g = hidden_g @ Wg
// ---------------------------------------------------------------------------
__global__ __launch_bounds__(256) void dn_gate_proj(
    const float* __restrict__ hab, const float* __restrict__ hg,
    const float* __restrict__ Wb, const float* __restrict__ Wg,
    float* __restrict__ beta_out, float* __restrict__ g_out) {
  using namespace dn;
  __shared__ _Float16 WbT[kH * kHID];  // [n][k], 64KB
  __shared__ _Float16 WgT[kH * kHID];  // [n][k], 64KB

  for (int i = threadIdx.x; i < kH * kHID; i += 256) {
    const int kk = i >> 4, n = i & 15;
    WbT[n * kHID + kk] = (_Float16)Wb[i];
    WgT[n * kHID + kk] = (_Float16)Wg[i];
  }
  __syncthreads();

  const int lane = threadIdx.x & 31;
  const int wave = threadIdx.x >> 5;
  const int row0 = (blockIdx.x * 8 + wave) * 16;

  v8f accB = {0.f, 0.f, 0.f, 0.f, 0.f, 0.f, 0.f, 0.f};
  v8f accG = {0.f, 0.f, 0.f, 0.f, 0.f, 0.f, 0.f, 0.f};
  for (int k0 = 0; k0 < kHID; k0 += 32) {
    const v16h aB = fragA_f32g(hab, kHID, row0, k0, lane);
    const v16h aG = fragA_f32g(hg, kHID, row0, k0, lane);
    const v16h bB = fragBT_lds(WbT, kHID, k0, 0, lane);
    const v16h bG = fragBT_lds(WgT, kHID, k0, 0, lane);
    accB = WMMA_F16(aB, bB, accB);
    accG = WMMA_F16(aG, bG, accG);
  }
  const int n = lane & 15, mo = (lane & 16) ? 8 : 0;
#pragma unroll
  for (int r = 0; r < 8; ++r) {
    const int row = row0 + r + mo;
    beta_out[row * kH + n] = sigmoidf_(accB[r]);
    g_out[row * kH + n] = accG[r];
  }
}

// ---------------------------------------------------------------------------
// Kernel 2: per-(b,h) chunked delta-rule scan. 32 workgroups x 256 threads.
// LDS layout (160KB exactly): ShT 32K | PhT 32K | u2(qh/oh) 16K | kh 16K |
// vbh 16K | u0(Af/wh/TDM-stage) 16K | u1(kbh/vih) 16K | Th 8K | attnh 8K.
// ---------------------------------------------------------------------------
__global__ __launch_bounds__(256) void dn_delta_chunk(
    const float* __restrict__ q, const float* __restrict__ k,
    const float* __restrict__ v, const float* __restrict__ betaw,
    const float* __restrict__ gw, const float* __restrict__ onw,
    const float* __restrict__ oproj, float* __restrict__ out) {
  using namespace dn;
  __shared__ _Float16 ShT[kDV * kDK];              // 32KB f16 state, ShT[v][d]
  __shared__ _Float16 PhT[kDV * kDK];              // 32KB o_proj f16, PhT[d][v]
  __shared__ __align__(16) char u2[kC * kDK * 2];  // 16KB: qh then oh
  __shared__ _Float16 kh[kC * kDK];                // 16KB  l2norm(k)
  __shared__ _Float16 vbh[kC * kDV];               // 16KB  v * beta
  __shared__ __align__(16) char u0[kC * kC * 4];   // 16KB: TDM stage / Af / wh
  __shared__ __align__(16) char u1[kC * kDK * 2];  // 16KB: kbh then vih
  __shared__ _Float16 Th[kC * kC];                 // 8KB (I+L)^-1; head = rowtmp
  __shared__ _Float16 attnh[kC * kC];              // 8KB tril(q k^T)

  _Float16* qh = reinterpret_cast<_Float16*>(u2);
  _Float16* oh = reinterpret_cast<_Float16*>(u2);
  float* Af = reinterpret_cast<float*>(u0);
  _Float16* wh = reinterpret_cast<_Float16*>(u0);
  float* stage = reinterpret_cast<float*>(u0);
  _Float16* kbh = reinterpret_cast<_Float16*>(u1);
  _Float16* vih = reinterpret_cast<_Float16*>(u1);
  float* rowtmp = reinterpret_cast<float*>(Th);    // alive only during inversion

  const int tid = threadIdx.x, lane = tid & 31, wave = tid >> 5;
  const int bIdx = blockIdx.x >> 4;  // H = 16
  const int hIdx = blockIdx.x & 15;

  const v8f vzero = {0.f, 0.f, 0.f, 0.f, 0.f, 0.f, 0.f, 0.f};
  v8f Sreg[8];                       // f32 state master: tiles (tm=j, tn=wave)
#pragma unroll
  for (int j = 0; j < 8; ++j) Sreg[j] = vzero;
  for (int i = tid; i < kDK * kDV; i += 256) ShT[i] = (_Float16)0.f;

  // ---- one-time: stage o_proj_w[h] as transposed f16 (TDM DMA, 4 slabs) ---
  const float* Ph = oproj + (size_t)hIdx * kDV * kDK;
#if defined(DN_HAS_TDM)
  for (int r0 = 0; r0 < kDV; r0 += 32) {
    __syncthreads();
    if (wave == 0)
      tdm_load_2d_f32((unsigned)(uintptr_t)(void*)stage, Ph + r0 * kDK, kDK, 32, kDK);
    __syncthreads();
    for (int i = tid; i < 32 * kDK; i += 256) {
      const int rr = i >> 7, cc = i & 127;
      PhT[cc * kDK + (r0 + rr)] = (_Float16)stage[i];
    }
  }
#else
  for (int i = tid; i < kDV * kDK; i += 256) {
    const int vv = i >> 7, dd = i & 127;
    PhT[dd * kDK + vv] = (_Float16)Ph[i];
  }
#endif
  __syncthreads();

  const int row = tid >> 2;   // 0..63 chunk row
  const int seg = tid & 3;    // 4 threads per row
  const int c0 = seg * 32;    // 32 contiguous feature cols each

  for (int chunk = 0; chunk < kN; ++chunk) {
    const int tglob = chunk * kC + row;
    const size_t tokBase = (size_t)bIdx * kT + tglob;
    const float beta = betaw[tokBase * kH + hIdx];

    // ---------------- A: load + l2norm q/k, beta-scale k/v ----------------
    {
      float t[32];
      const float* qp = q + (tokBase * kH + hIdx) * (size_t)kDK + c0;
      float ss = 0.f;
#pragma unroll
      for (int i = 0; i < 8; ++i) {
        const float4 f = *reinterpret_cast<const float4*>(qp + i * 4);
        t[4 * i] = f.x; t[4 * i + 1] = f.y; t[4 * i + 2] = f.z; t[4 * i + 3] = f.w;
        ss += f.x * f.x + f.y * f.y + f.z * f.z + f.w * f.w;
      }
      ss += __shfl_xor(ss, 1); ss += __shfl_xor(ss, 2);
      const float rnq = rsqrtf(ss + 1e-6f) * kScale;
#pragma unroll
      for (int j = 0; j < 32; ++j) qh[row * kDK + c0 + j] = (_Float16)(t[j] * rnq);

      const float* kp = k + (tokBase * kH + hIdx) * (size_t)kDK + c0;
      ss = 0.f;
#pragma unroll
      for (int i = 0; i < 8; ++i) {
        const float4 f = *reinterpret_cast<const float4*>(kp + i * 4);
        t[4 * i] = f.x; t[4 * i + 1] = f.y; t[4 * i + 2] = f.z; t[4 * i + 3] = f.w;
        ss += f.x * f.x + f.y * f.y + f.z * f.z + f.w * f.w;
      }
      ss += __shfl_xor(ss, 1); ss += __shfl_xor(ss, 2);
      const float rnk = rsqrtf(ss + 1e-6f);
#pragma unroll
      for (int j = 0; j < 32; ++j) {
        const float kn = t[j] * rnk;
        kh[row * kDK + c0 + j] = (_Float16)kn;
        kbh[row * kDK + c0 + j] = (_Float16)(kn * beta);
      }

      const float* vp = v + (tokBase * kH + hIdx) * (size_t)kDV + c0;
#pragma unroll
      for (int i = 0; i < 8; ++i) {
        const float4 f = *reinterpret_cast<const float4*>(vp + i * 4);
        vbh[row * kDV + c0 + 4 * i]     = (_Float16)(f.x * beta);
        vbh[row * kDV + c0 + 4 * i + 1] = (_Float16)(f.y * beta);
        vbh[row * kDV + c0 + 4 * i + 2] = (_Float16)(f.z * beta);
        vbh[row * kDV + c0 + 4 * i + 3] = (_Float16)(f.w * beta);
      }
    }
    __syncthreads();

    // ---------- B: Af = -strict_tril(kb k^T); attn = tril(q k^T) ----------
    for (int t8 = wave; t8 < 32; t8 += 8) {
      const bool isA = t8 < 16;
      const int tt = t8 & 15, tm = tt >> 2, tn = tt & 3;
      v8f acc = vzero;
      for (int k0 = 0; k0 < kDK; k0 += 32) {
        const v16h a = fragA_lds(isA ? kbh : qh, kDK, tm * 16, k0, lane);
        const v16h b = fragBT_lds(kh, kDK, k0, tn * 16, lane);
        acc = WMMA_F16(a, b, acc);
      }
      const int n = tn * 16 + (lane & 15), mo = (lane & 16) ? 8 : 0;
#pragma unroll
      for (int r = 0; r < 8; ++r) {
        const int m = tm * 16 + r + mo;
        if (isA) Af[m * kC + n] = (m > n) ? -acc[r] : 0.f;
        else     attnh[m * kC + n] = (_Float16)((m >= n) ? acc[r] : 0.f);
      }
    }
    __syncthreads();

    // ---------- C: forward substitution (single wave, barrier-free) -------
    if (wave == 0) {
      for (int i = 1; i < kC; ++i) {
        rowtmp[lane] = Af[i * kC + lane];
        rowtmp[lane + 32] = Af[i * kC + lane + 32];
        float s0 = rowtmp[lane];
        float s1 = rowtmp[lane + 32];
        for (int j = 0; j < i; ++j) {
          const float rj = rowtmp[j];
          s0 += rj * Af[j * kC + lane];
          s1 += rj * Af[j * kC + lane + 32];
        }
        Af[i * kC + lane] = s0;
        Af[i * kC + lane + 32] = s1;
      }
    }
    __syncthreads();
    for (int i = tid; i < kC * kC; i += 256) {
      const int r = i >> 6, c = i & 63;
      Th[i] = (c < r) ? (_Float16)Af[i] : (_Float16)((c == r) ? 1.f : 0.f);
    }
    __syncthreads();

    // ---------- D: w = T @ k_beta  (writes wh over dead Af) ---------------
#pragma unroll
    for (int j = 0; j < 4; ++j) {
      v8f acc = vzero;
#pragma unroll
      for (int k0 = 0; k0 < kC; k0 += 32)
        acc = WMMA_F16(fragA_lds(Th, kC, j * 16, k0, lane),
                       fragB_lds(kbh, kDK, k0, wave * 16, lane), acc);
      storeD_f16(wh, kDK, j * 16, wave * 16, lane, acc);
    }
    __syncthreads();

    // ---------- E: OAcc = q@S ; vih = T@v_beta - w@S ----------------------
    v8f OAcc[4];
#pragma unroll
    for (int j = 0; j < 4; ++j) {
      v8f acc = vzero;
#pragma unroll
      for (int k0 = 0; k0 < kDK; k0 += 32)
        acc = WMMA_F16(fragA_lds(qh, kDK, j * 16, k0, lane),
                       fragBT_lds(ShT, kDK, k0, wave * 16, lane), acc);
      OAcc[j] = acc;
    }
#pragma unroll
    for (int j = 0; j < 4; ++j) {
      v8f acc = vzero;
#pragma unroll
      for (int k0 = 0; k0 < kDK; k0 += 32)
        acc = WMMA_F16(fragA_lds(wh, kDK, j * 16, k0, lane),
                       fragBT_lds(ShT, kDK, k0, wave * 16, lane), acc);
#pragma unroll
      for (int r = 0; r < 8; ++r) acc[r] = -acc[r];
#pragma unroll
      for (int k0 = 0; k0 < kC; k0 += 32)
        acc = WMMA_F16(fragA_lds(Th, kC, j * 16, k0, lane),
                       fragB_lds(vbh, kDV, k0, wave * 16, lane), acc);
      storeD_f16(vih, kDV, j * 16, wave * 16, lane, acc);  // over dead kbh
    }
    __syncthreads();

    // ---------- F: OAcc += attn @ vi ; spill o (over dead qh) -------------
#pragma unroll
    for (int j = 0; j < 4; ++j) {
      v8f acc = OAcc[j];
#pragma unroll
      for (int k0 = 0; k0 < kC; k0 += 32)
        acc = WMMA_F16(fragA_lds(attnh, kC, j * 16, k0, lane),
                       fragB_lds(vih, kDV, k0, wave * 16, lane), acc);
      storeD_f16(oh, kDV, j * 16, wave * 16, lane, acc);
    }

    // ---------- G: Sreg += k^T @ vi ; refresh f16 mirror ShT --------------
#pragma unroll
    for (int j = 0; j < 8; ++j) {
      v8f acc = Sreg[j];
#pragma unroll
      for (int k0 = 0; k0 < kC; k0 += 32)
        acc = WMMA_F16(fragAT_lds(kh, kDK, j * 16, k0, lane),
                       fragB_lds(vih, kDV, k0, wave * 16, lane), acc);
      Sreg[j] = acc;
      storeD_f16T(ShT, kDK, j * 16, wave * 16, lane, acc);
    }
    __syncthreads();

    // ---------- H: RMSNorm * o_norm_w * swish(g) per row ------------------
    {
      float t[32];
      float ss = 0.f;
#pragma unroll
      for (int j = 0; j < 32; ++j) {
        const float x = (float)oh[row * kDV + c0 + j];
        t[j] = x; ss += x * x;
      }
      ss += __shfl_xor(ss, 1); ss += __shfl_xor(ss, 2);
      const float sc = rsqrtf(ss * (1.f / kDV) + 1e-5f);
      const float gv = gw[tokBase * kH + hIdx];
      const float gate = gv * sigmoidf_(gv);
#pragma unroll
      for (int j = 0; j < 32; ++j)
        oh[row * kDV + c0 + j] = (_Float16)(t[j] * sc * onw[c0 + j] * gate);
    }
    __syncthreads();

    // ---------- I: out = gated_o @ PhT (LDS f16, contiguous B reads) ------
#pragma unroll
    for (int j = 0; j < 4; ++j) {
      v8f acc = vzero;
#pragma unroll
      for (int k0 = 0; k0 < kDV; k0 += 32)
        acc = WMMA_F16(fragA_lds(oh, kDV, j * 16, k0, lane),
                       fragBT_lds(PhT, kDK, k0, wave * 16, lane), acc);
      const int n = wave * 16 + (lane & 15), mo = (lane & 16) ? 8 : 0;
#pragma unroll
      for (int r = 0; r < 8; ++r) {
        const int m = j * 16 + r + mo;
        out[((size_t)bIdx * kT + chunk * kC + m) * (kH * kDK) + hIdx * kDK + n] = acc[r];
      }
    }

    // prefetch next chunk's q/k/v rows into cache (global_prefetch_b8)
    if (chunk + 1 < kN) {
      const size_t nb = ((size_t)bIdx * kT + (chunk + 1) * kC + row) * kH + hIdx;
      __builtin_prefetch(q + nb * kDK + c0, 0, 1);
      __builtin_prefetch(k + nb * kDK + c0, 0, 1);
      __builtin_prefetch(v + nb * kDV + c0, 0, 1);
    }
    __syncthreads();
  }
}

// ---------------------------------------------------------------------------
extern "C" void kernel_launch(void* const* d_in, const int* in_sizes, int n_in,
                              void* d_out, int out_size, void* d_ws, size_t ws_size,
                              hipStream_t stream) {
  (void)in_sizes; (void)n_in; (void)out_size; (void)ws_size;
  using namespace dn;
  const float* hab = (const float*)d_in[0];
  const float* hg  = (const float*)d_in[1];
  const float* q   = (const float*)d_in[2];
  const float* k   = (const float*)d_in[3];
  const float* v   = (const float*)d_in[4];
  const float* Wb  = (const float*)d_in[5];
  const float* Wg  = (const float*)d_in[6];
  const float* onw = (const float*)d_in[7];
  const float* op  = (const float*)d_in[8];
  float* out = (float*)d_out;

  float* beta_ws = (float*)d_ws;                    // [B*T*H]
  float* g_ws    = beta_ws + (size_t)kB * kT * kH;  // [B*T*H]

  dn_gate_proj<<<dim3((kB * kT) / 128), dim3(256), 0, stream>>>(
      hab, hg, Wb, Wg, beta_ws, g_ws);
  dn_delta_chunk<<<dim3(kB * kH), dim3(256), 0, stream>>>(
      q, k, v, beta_ws, g_ws, onw, op, out);
}
